// NonLocalBlock_85521388798563
// MI455X (gfx1250) — matrix operand order repaired and checked by reading, not compiled, not run on hardware
//
#include <hip/hip_runtime.h>

// ---- problem constants -------------------------------------------------
#define BB   2
#define INC  64
#define NC   32
#define HH   96
#define WW   96
#define HO   94
#define WO   94
#define NN   (HO * WO)            // 8836
#define NT   ((NN + 15) / 16)     // 553 n/m tiles of 16
#define MC2  ((NN + 31) / 32)     // 277 m chunks of 32
#define EPS  1e-5f

typedef __attribute__((ext_vector_type(16))) _Float16 v16h;
typedef __attribute__((ext_vector_type(8)))  _Float16 v8h;
typedef __attribute__((ext_vector_type(8)))  float    v8f;

// ---- kernel 1: 3x3 VALID conv + bias + BN + LeakyReLU -> f16 [B][N][32]
__global__ void convbn_kernel(const float* __restrict__ x,
                              const float* __restrict__ w,
                              const float* __restrict__ bias,
                              const float* __restrict__ g,
                              const float* __restrict__ be,
                              const float* __restrict__ m,
                              const float* __restrict__ v,
                              _Float16* __restrict__ out)
{
    int t = blockIdx.x * blockDim.x + threadIdx.x;
    if (t >= BB * NN * NC) return;
    int c = t & 31;
    int n = (t >> 5) % NN;
    int b = t / (NN * NC);
    int p = n / WO, q = n % WO;

    const float* xb = x + ((size_t)b * INC) * (HH * WW) + p * WW + q;
    const float* wc = w + (size_t)c * INC * 9;
    float acc = 0.f;
    for (int ic = 0; ic < INC; ++ic) {
        const float* xp = xb + (size_t)ic * (HH * WW);
        const float* wp = wc + ic * 9;
        #pragma unroll
        for (int kh = 0; kh < 3; ++kh)
            #pragma unroll
            for (int kw = 0; kw < 3; ++kw)
                acc += xp[kh * WW + kw] * wp[kh * 3 + kw];
    }
    float scale = g[c] * rsqrtf(v[c] + EPS);
    float y = (acc + bias[c] - m[c]) * scale + be[c];
    y = y > 0.f ? y : 0.01f * y;
    out[t] = (_Float16)y;   // t == ((b*NN + n)*32 + c)
}

// ---- kernel 2: column softmax stats  M[m]=max_n s, Z[m]=sum_n exp(s-M)
// one wave per 16-column m-tile; loops n-tiles; 1 v_wmma per iteration
__global__ void __launch_bounds__(32) attn_stats_kernel(
    const _Float16* __restrict__ Q, const _Float16* __restrict__ K,
    float* __restrict__ Mst, float* __restrict__ Zst)
{
    int b  = blockIdx.x / NT;
    int mt = blockIdx.x % NT;
    int lane = threadIdx.x;
    int half = lane >> 4;           // 0 or 1
    int l16  = lane & 15;

    const _Float16* Qb = Q + (size_t)b * NN * NC;
    const _Float16* Kb = K + (size_t)b * NN * NC;

    // B fragment: lane holds K-row (m0+l16), k = 16*half + 0..15 (contiguous)
    int mrow = mt * 16 + l16; if (mrow >= NN) mrow = NN - 1;
    v16h bfrag = *(const v16h*)(Kb + (size_t)mrow * NC + 16 * half);

    float runM = -3.0e38f, runZ = 0.f;
    for (int nt = 0; nt < NT; ++nt) {
        int n0 = nt * 16;
        int nrow = n0 + l16; if (nrow >= NN) nrow = NN - 1;
        const _Float16* qr = Qb + (size_t)nrow * NC;
        v16h a;
        a = *(const v16h*)(qr + 16 * half);                      // k = 16h..16h+15
        v8f acc = {};
        acc = __builtin_amdgcn_wmma_f32_16x16x32_f16(
                  false, a, false, bfrag, (short)0, acc, false, false);
        // D layout: reg j -> row (j + 8*half), col = l16
        float tm = -3.0e38f;
        #pragma unroll
        for (int j = 0; j < 8; ++j) {
            int n = n0 + j + 8 * half;
            if (n < NN) tm = fmaxf(tm, acc[j]);
        }
        tm = fmaxf(tm, __shfl_xor(tm, 16, 32));      // merge other 8 rows
        float nm = fmaxf(runM, tm);
        float ts = 0.f;
        #pragma unroll
        for (int j = 0; j < 8; ++j) {
            int n = n0 + j + 8 * half;
            if (n < NN) ts += __expf(acc[j] - nm);
        }
        ts += __shfl_xor(ts, 16, 32);
        runZ = runZ * __expf(runM - nm) + ts;        // online rescale
        runM = nm;
    }
    int mg = mt * 16 + l16;
    if (half == 0 && mg < NN) {
        Mst[(size_t)b * NN + mg] = runM;
        Zst[(size_t)b * NN + mg] = runZ;
    }
}

// ---- kernel 3: o[n,c] = sum_m exp(s[n,m]-M[m])/Z[m] * V[m,c]
// one wave per 16-row n-tile; 32 m per iter: 2 score WMMAs + 2 accum WMMAs
__global__ void __launch_bounds__(32) attn_out_kernel(
    const _Float16* __restrict__ Q, const _Float16* __restrict__ K,
    const _Float16* __restrict__ V,
    const float* __restrict__ Mst, const float* __restrict__ Zst,
    float* __restrict__ O)
{
    __shared__ __align__(32) _Float16 Pl[16 * 32];   // P tile, row-major

    int b  = blockIdx.x / NT;
    int nt = blockIdx.x % NT;
    int lane = threadIdx.x;
    int half = lane >> 4;
    int l16  = lane & 15;

    const _Float16* Qb = Q + (size_t)b * NN * NC;
    const _Float16* Kb = K + (size_t)b * NN * NC;
    const _Float16* Vb = V + (size_t)b * NN * NC;
    const float*    Mb = Mst + (size_t)b * NN;
    const float*    Zb = Zst + (size_t)b * NN;

    int nrow = nt * 16 + l16; if (nrow >= NN) nrow = NN - 1;
    const _Float16* qr = Qb + (size_t)nrow * NC;
    v16h aQ;
    aQ = *(const v16h*)(qr + 16 * half);

    v8f oacc0 = {};   // c = 0..15
    v8f oacc1 = {};   // c = 16..31

    for (int mc = 0; mc < MC2; ++mc) {
        int m0 = mc * 32;
        v8f s[2];
        #pragma unroll
        for (int t = 0; t < 2; ++t) {
            int mrow = m0 + 16 * t + l16; if (mrow >= NN) mrow = NN - 1;
            v16h bf = *(const v16h*)(Kb + (size_t)mrow * NC + 16 * half);
            v8f z = {};
            s[t] = __builtin_amdgcn_wmma_f32_16x16x32_f16(
                       false, aQ, false, bf, (short)0, z, false, false);
        }
        // P = exp(s - M[m]) / Z[m]  (per-COLUMN stats), masked past N
        #pragma unroll
        for (int t = 0; t < 2; ++t) {
            int mg = m0 + 16 * t + l16;
            float Mv = 0.f, Zi = 0.f;
            if (mg < NN) { Mv = Mb[mg]; Zi = 1.0f / Zb[mg]; }
            #pragma unroll
            for (int j = 0; j < 8; ++j) {
                int r = j + 8 * half;
                float pp = (mg < NN) ? __expf(s[t][j] - Mv) * Zi : 0.f;
                Pl[r * 32 + 16 * t + l16] = (_Float16)pp;
            }
        }
        __syncthreads();
        // P as A fragment (rows n, K = local m 0..31)
        const _Float16* pr = Pl + l16 * 32;
        v16h aP;
        aP = *(const v16h*)(pr + 16 * half);
        // V as B fragments: element i -> V[m0 + 16*half + i][c0 + l16]
        v16h bv0, bv1;
        #pragma unroll
        for (int i = 0; i < 16; ++i) {
            int mr = m0 + 16 * half + i; if (mr >= NN) mr = NN - 1;
            const _Float16* vr = Vb + (size_t)mr * NC;
            bv0[i] = vr[l16];
            bv1[i] = vr[16 + l16];
        }
        oacc0 = __builtin_amdgcn_wmma_f32_16x16x32_f16(
                    false, aP, false, bv0, (short)0, oacc0, false, false);
        oacc1 = __builtin_amdgcn_wmma_f32_16x16x32_f16(
                    false, aP, false, bv1, (short)0, oacc1, false, false);
        __syncthreads();
    }

    float* Ob = O + (size_t)b * NN * NC;
    #pragma unroll
    for (int j = 0; j < 8; ++j) {
        int n = nt * 16 + j + 8 * half;
        if (n < NN) {
            Ob[(size_t)n * NC + l16]      = oacc0[j];
            Ob[(size_t)n * NC + 16 + l16] = oacc1[j];
        }
    }
}

// ---- kernel 4: ConvTranspose 3x3 VALID + bias + BN + LeakyReLU + residual
__global__ void convt_kernel(const float* __restrict__ x,
                             const float* __restrict__ O,
                             const float* __restrict__ wt,
                             const float* __restrict__ bt,
                             const float* __restrict__ g4,
                             const float* __restrict__ be4,
                             const float* __restrict__ m4,
                             const float* __restrict__ v4,
                             float* __restrict__ out)
{
    int t = blockIdx.x * blockDim.x + threadIdx.x;
    if (t >= BB * INC * HH * WW) return;
    int j  = t % WW;
    int i  = (t / WW) % HH;
    int oc = (t / (HH * WW)) % INC;
    int b  = t / (HH * WW * INC);

    float acc = 0.f;
    #pragma unroll
    for (int kh = 0; kh < 3; ++kh) {
        int p = i - kh; if (p < 0 || p >= HO) continue;
        #pragma unroll
        for (int kw = 0; kw < 3; ++kw) {
            int q = j - kw; if (q < 0 || q >= WO) continue;
            const float* op = O + ((size_t)b * NN + (size_t)p * WO + q) * NC;
            const float* wp = wt + oc * 9 + kh * 3 + kw;  // wt[c][oc][kh][kw]
            #pragma unroll
            for (int c = 0; c < NC; ++c)
                acc += op[c] * wp[(size_t)c * INC * 9];
        }
    }
    float scale = g4[oc] * rsqrtf(v4[oc] + EPS);
    float y = (acc + bt[oc] - m4[oc]) * scale + be4[oc];
    y = y > 0.f ? y : 0.01f * y;
    out[t] = x[t] + y;
}

// ---- launch ------------------------------------------------------------
extern "C" void kernel_launch(void* const* d_in, const int* in_sizes, int n_in,
                              void* d_out, int out_size, void* d_ws, size_t ws_size,
                              hipStream_t stream)
{
    (void)in_sizes; (void)n_in; (void)out_size; (void)ws_size;
    const float* x  = (const float*)d_in[0];
    const float* wt = (const float*)d_in[19];
    const float* bt = (const float*)d_in[20];
    const float* g4 = (const float*)d_in[21];
    const float* be4= (const float*)d_in[22];
    const float* m4 = (const float*)d_in[23];
    const float* v4 = (const float*)d_in[24];

    // workspace layout (~5.8 MB): q/k/v f16 [B][N][32], o f32 [B][N][32], M/Z
    char* ws = (char*)d_ws;
    const size_t QKVB = (size_t)BB * NN * NC * sizeof(_Float16); // 1,131,008
    const size_t OB   = (size_t)BB * NN * NC * sizeof(float);    // 2,262,016
    _Float16* qh = (_Float16*)(ws);
    _Float16* kh = (_Float16*)(ws + QKVB);
    _Float16* vh = (_Float16*)(ws + 2 * QKVB);
    float*    Oo = (float*)(ws + 3 * QKVB);
    float*    Ms = (float*)(ws + 3 * QKVB + OB);
    float*    Zs = (float*)(ws + 3 * QKVB + OB + (size_t)BB * NN * sizeof(float));

    const int cbThreads = 256;
    const int cbBlocks  = (BB * NN * NC + cbThreads - 1) / cbThreads; // 2209
    _Float16* dst[3] = { qh, kh, vh };
    for (int br = 0; br < 3; ++br) {
        const float* w  = (const float*)d_in[1 + 6 * br + 0];
        const float* bb = (const float*)d_in[1 + 6 * br + 1];
        const float* g  = (const float*)d_in[1 + 6 * br + 2];
        const float* be = (const float*)d_in[1 + 6 * br + 3];
        const float* m  = (const float*)d_in[1 + 6 * br + 4];
        const float* v  = (const float*)d_in[1 + 6 * br + 5];
        convbn_kernel<<<cbBlocks, cbThreads, 0, stream>>>(x, w, bb, g, be, m, v, dst[br]);
    }

    attn_stats_kernel<<<BB * NT, 32, 0, stream>>>(qh, kh, Ms, Zs);
    attn_out_kernel  <<<BB * NT, 32, 0, stream>>>(qh, kh, vh, Ms, Zs, Oo);

    const int ctBlocks = (BB * INC * HH * WW + 255) / 256; // 4608
    convt_kernel<<<ctBlocks, 256, 0, stream>>>(x, Oo, wt, bt, g4, be4, m4, v4,
                                               (float*)d_out);
}
